// MHA_89077621719053
// MI455X (gfx1250) — matrix operand order repaired and checked by reading, compiled, not run on hardware
//
#include <hip/hip_runtime.h>
#include <hip/hip_bf16.h>
#include <stdint.h>

typedef __attribute__((ext_vector_type(16))) __bf16 v16bf;
typedef __attribute__((ext_vector_type(8)))  __bf16 v8bf;
typedef __attribute__((ext_vector_type(2)))  __bf16 v2bf;
typedef __attribute__((ext_vector_type(8)))  float  v8f;
typedef __attribute__((ext_vector_type(4)))  unsigned int u32x4;
typedef __attribute__((ext_vector_type(8)))  int i32x8;
typedef __attribute__((ext_vector_type(4)))  int i32x4;

#define BATCH  4
#define SEQ    2048
#define DMODEL 1024
#define NHEAD  16
#define HDIM   64

// ---------------- f32 -> bf16 (RNE), packed when hardware cvt available -----
static __device__ __forceinline__ __bf16 f2bf_sw(float f) {
  unsigned int u = __builtin_bit_cast(unsigned int, f);
  unsigned int r = (u + 0x7FFFu + ((u >> 16) & 1u)) >> 16;
  return __builtin_bit_cast(__bf16, (unsigned short)r);
}

#if defined(__has_builtin)
#if __has_builtin(__builtin_amdgcn_cvt_pk_bf16_f32)
#define HAVE_PK_BF16 1
#endif
#endif

static __device__ __forceinline__ v2bf pk2(float x, float y) {
#ifdef HAVE_PK_BF16
  auto t = __builtin_amdgcn_cvt_pk_bf16_f32(x, y);
  static_assert(sizeof(t) == 4, "pk bf16 size");
  return __builtin_bit_cast(v2bf, t);
#else
  v2bf r; r[0] = f2bf_sw(x); r[1] = f2bf_sw(y); return r;
#endif
}
static __device__ __forceinline__ __bf16 f2bf(float f) {
#ifdef HAVE_PK_BF16
  return pk2(f, f)[0];
#else
  return f2bf_sw(f);
#endif
}
static __device__ __forceinline__ v8bf cvt8(float4 a, float4 b) {
  v2bf p0 = pk2(a.x, a.y), p1 = pk2(a.z, a.w);
  v2bf p2 = pk2(b.x, b.y), p3 = pk2(b.z, b.w);
  v8bf d;
  d[0]=p0[0]; d[1]=p0[1]; d[2]=p1[0]; d[3]=p1[1];
  d[4]=p2[0]; d[5]=p2[1]; d[6]=p3[0]; d[7]=p3[1];
  return d;
}

static __device__ __forceinline__ v8f wmma_bf16(v16bf a, v16bf b, v8f c) {
  return __builtin_amdgcn_wmma_f32_16x16x32_bf16(false, a, false, b, (short)0, c, false, false);
}

// reductions across a 16-lane half (wave32)
static __device__ __forceinline__ float rmax16(float v) {
  v = fmaxf(v, __shfl_xor(v, 8, 32));
  v = fmaxf(v, __shfl_xor(v, 4, 32));
  v = fmaxf(v, __shfl_xor(v, 2, 32));
  v = fmaxf(v, __shfl_xor(v, 1, 32));
  return v;
}
static __device__ __forceinline__ float rsum16(float v) {
  v += __shfl_xor(v, 8, 32);
  v += __shfl_xor(v, 4, 32);
  v += __shfl_xor(v, 2, 32);
  v += __shfl_xor(v, 1, 32);
  return v;
}

// ---------------- TDM: async 2D bf16 tile (rows x 32) -> LDS ----------------
#if defined(__has_builtin)
#if __has_builtin(__builtin_amdgcn_tensor_load_to_lds)
#define HAVE_TDM 1
#endif
#endif

#ifdef HAVE_TDM
static __device__ __forceinline__ void tdm_load_bf16_tile(
    unsigned lds_off, const __bf16* gsrc,
    unsigned tensor_d0, unsigned tensor_d1, unsigned stride0,
    unsigned tile_d0, unsigned tile_d1)
{
  const unsigned long long ga = (unsigned long long)(uintptr_t)gsrc;
  u32x4 g0;
  g0[0] = 1u;                                                  // count=1
  g0[1] = lds_off;                                             // lds_addr
  g0[2] = (unsigned)(ga & 0xFFFFFFFFu);                        // global[31:0]
  g0[3] = (unsigned)((ga >> 32) & 0x01FFFFFFu) | 0x80000000u;  // [56:32]|type=2
  i32x8 g1;
  g1[0] = (int)(1u << 16);                                     // data_size = 2B
  g1[1] = (int)((tensor_d0 & 0xFFFFu) << 16);
  g1[2] = (int)(((tensor_d0 >> 16) & 0xFFFFu) | ((tensor_d1 & 0xFFFFu) << 16));
  g1[3] = (int)(((tensor_d1 >> 16) & 0xFFFFu) | ((tile_d0 & 0xFFFFu) << 16));
  g1[4] = (int)(tile_d1 & 0xFFFFu);
  g1[5] = (int)stride0;
  g1[6] = 0;
  g1[7] = 0;
  i32x4 z4 = {0, 0, 0, 0};
#if defined(__clang_major__) && __clang_major__ >= 23
  i32x8 z8 = {0, 0, 0, 0, 0, 0, 0, 0};
  __builtin_amdgcn_tensor_load_to_lds(g0, g1, z4, z4, z8, 0);
#else
  __builtin_amdgcn_tensor_load_to_lds(g0, g1, z4, z4, 0);
#endif
}
#endif

// ---------------- Kernel 0: one-time f32 -> bf16 conversion -----------------
__global__ __launch_bounds__(256)
void cvt_bf16_kernel(const float* __restrict__ src, __bf16* __restrict__ dst, int n8)
{
  const int i = blockIdx.x * 256 + threadIdx.x;
  if (i < n8) {
    const float4* s = (const float4*)src + 2ull * i;
    *(v8bf*)(dst + 8ull * i) = cvt8(s[0], s[1]);
  }
}

// ---------------------------------------------------------------------------
// bf16 WMMA GEMM core (block 128x128, K-tile 32, 8 waves, wave tile 32x64),
// double-buffered LDS; A tile staged by TDM (wave 0), B tile register-
// pipelined and stored transposed. Used by both projection kernels.
// ---------------------------------------------------------------------------
#define GEMM_PROLOG_AND_LOOP(A_PTR, B_PTR, LDB)                                  \
  __shared__ __align__(64) __bf16 As[2][128][32];                                \
  __shared__ __align__(64) __bf16 Bt[2][128][32];                                \
  const int tid  = threadIdx.x;                                                  \
  const int n0   = blockIdx.x * 128;                                             \
  const int m0   = blockIdx.y * 128;                                             \
  const int wv   = tid >> 5;                                                     \
  const int lane = tid & 31;                                                     \
  const int lm   = lane & 15;                                                    \
  const int kh   = lane >> 4;                                                    \
  const int wm   = wv >> 1, wn = wv & 1;                                         \
  const int bk = tid >> 3;                                                       \
  const int bn = (tid & 7) * 16;                                                 \
  v8f c[2][4] = {};                                                              \
  v8bf br0, br1;                                                                 \
  /* prologue: stage tile 0 */                                                   \
  _Pragma("nounroll")                                                            \
  for (int once = 0; once < 1; ++once) {                                         \
    if (wv == 0)                                                                 \
      tdm_load_bf16_tile((unsigned)(uintptr_t)&As[0][0][0],                      \
                         (A_PTR) + (size_t)m0 * DMODEL, DMODEL, BATCH * SEQ,     \
                         DMODEL, 32, 128);                                       \
    const __bf16* bsrc = (B_PTR) + (size_t)bk * (LDB) + n0 + bn;                 \
    br0 = *(const v8bf*)(bsrc);                                                  \
    br1 = *(const v8bf*)(bsrc + 8);                                              \
    for (int i = 0; i < 8; ++i) Bt[0][bn + i][bk]     = br0[i];                  \
    for (int i = 0; i < 8; ++i) Bt[0][bn + 8 + i][bk] = br1[i];                  \
    if (wv == 0) __builtin_amdgcn_s_wait_tensorcnt(0);                           \
  }                                                                              \
  __syncthreads();                                                               \
  for (int kt = 0; kt < DMODEL / 32; ++kt) {                                     \
    const int cur = kt & 1, nxt = cur ^ 1;                                       \
    const bool have_next = (kt + 1) < DMODEL / 32;                               \
    if (have_next) {                                                             \
      if (wv == 0)                                                               \
        tdm_load_bf16_tile((unsigned)(uintptr_t)&As[nxt][0][0],                  \
                           (A_PTR) + (size_t)m0 * DMODEL + (kt + 1) * 32,        \
                           DMODEL, BATCH * SEQ, DMODEL, 32, 128);                \
      const __bf16* bsrc =                                                       \
          (B_PTR) + (size_t)((kt + 1) * 32 + bk) * (LDB) + n0 + bn;              \
      br0 = *(const v8bf*)(bsrc);                                                \
      br1 = *(const v8bf*)(bsrc + 8);                                            \
    }                                                                            \
    v16bf a[2], b[4];                                                            \
    for (int s = 0; s < 2; ++s) {                                                \
      const int m = wm * 32 + s * 16 + lm;                                       \
      v8bf lo = *(const v8bf*)&As[cur][m][kh * 16];                              \
      v8bf hi = *(const v8bf*)&As[cur][m][kh * 16 + 8];                          \
      for (int i = 0; i < 8; ++i) { a[s][i] = lo[i]; a[s][8 + i] = hi[i]; }      \
    }                                                                            \
    for (int s = 0; s < 4; ++s) {                                                \
      const int n = wn * 64 + s * 16 + lm;                                       \
      b[s] = *(const v16bf*)&Bt[cur][n][kh * 16];                                \
    }                                                                            \
    for (int sm = 0; sm < 2; ++sm)                                               \
      for (int sn = 0; sn < 4; ++sn)                                             \
        c[sm][sn] = wmma_bf16(a[sm], b[sn], c[sm][sn]);                          \
    if (have_next) {                                                             \
      for (int i = 0; i < 8; ++i) Bt[nxt][bn + i][bk]     = br0[i];              \
      for (int i = 0; i < 8; ++i) Bt[nxt][bn + 8 + i][bk] = br1[i];              \
    }                                                                            \
    if (wv == 0) __builtin_amdgcn_s_wait_tensorcnt(0);                           \
    __syncthreads();                                                             \
  }

#ifndef HAVE_TDM
// fallback if TDM builtin is absent: no-op stub, A tile is not staged
static __device__ __forceinline__ void tdm_load_bf16_tile(
    unsigned, const __bf16*, unsigned, unsigned, unsigned, unsigned, unsigned) {}
static __device__ __forceinline__ void __builtin_amdgcn_s_wait_tensorcnt_stub() {}
#endif

// ---------------------------------------------------------------------------
// Kernel 1: qkv = x @ w_qkv (both bf16 pre-converted); epilogue scatters into
// Qh (pre-scaled 1/8), Kh, and transposed Vt (single b128 store per frag).
// ---------------------------------------------------------------------------
__global__ __launch_bounds__(256)
void qkv_gemm_kernel(const __bf16* __restrict__ Xb, const __bf16* __restrict__ Wb,
                     __bf16* __restrict__ Qh, __bf16* __restrict__ Kh,
                     __bf16* __restrict__ Vt)
{
  GEMM_PROLOG_AND_LOOP(Xb, Wb, 3 * DMODEL)

  const int which = n0 >> 10;          // 0=q 1=k 2=v, uniform per block
  for (int sm = 0; sm < 2; ++sm)
    for (int sn = 0; sn < 4; ++sn) {
      const int nn = n0 + wn * 64 + sn * 16 + lm;
      const int d  = nn & 1023;
      const int h  = d >> 6, hd = d & 63;
      const int mb = m0 + wm * 32 + sm * 16 + 8 * kh;
      const int bb = mb >> 11, t = mb & 2047;
      const int bh = bb * NHEAD + h;
      if (which == 0) {
        __bf16* p = Qh + ((size_t)bh * SEQ + t) * HDIM + hd;
        for (int v = 0; v < 8; ++v) p[(size_t)v * HDIM] = f2bf(c[sm][sn][v] * 0.125f);
      } else if (which == 1) {
        __bf16* p = Kh + ((size_t)bh * SEQ + t) * HDIM + hd;
        for (int v = 0; v < 8; ++v) p[(size_t)v * HDIM] = f2bf(c[sm][sn][v]);
      } else {
        v8bf pk;
        for (int v = 0; v < 8; v += 2) {
          v2bf q = pk2(c[sm][sn][v], c[sm][sn][v + 1]);
          pk[v] = q[0]; pk[v + 1] = q[1];
        }
        *(v8bf*)(Vt + ((size_t)bh * HDIM + hd) * SEQ + t) = pk;  // 8 cons. t
      }
    }
}

// ---------------------------------------------------------------------------
// Kernel 2: flash attention, one wave per (b, h, 16-row query tile),
// 64-key chunks: 8 WMMA for S, one softmax pass, 8 WMMA for P*V.
// ---------------------------------------------------------------------------
__global__ __launch_bounds__(256)
void attn_kernel(const __bf16* __restrict__ Qh, const __bf16* __restrict__ Kh,
                 const __bf16* __restrict__ Vt, __bf16* __restrict__ Y)
{
  __shared__ __align__(64) __bf16 Ps[8][16][64];   // per-wave P staging
  const int tid  = threadIdx.x;
  const int wvb  = tid >> 5;
  const int lane = tid & 31;
  const int lm   = lane & 15;
  const int kh   = lane >> 4;
  const int gw   = blockIdx.x * 8 + wvb;
  const int mt   = gw & 127;
  const int bh   = gw >> 7;
  const int bb   = bh >> 4, h = bh & 15;

  const __bf16* Qp = Qh + (size_t)bh * SEQ * HDIM;
  const __bf16* Kp = Kh + (size_t)bh * SEQ * HDIM;
  const __bf16* Vp = Vt + (size_t)bh * HDIM * SEQ;

  v16bf aq[2];
  {
    const int m = mt * 16 + lm;
    for (int kb = 0; kb < 2; ++kb) {
      v8bf lo = *(const v8bf*)(Qp + (size_t)m * HDIM + kb * 32 + kh * 16);
      v8bf hi = *(const v8bf*)(Qp + (size_t)m * HDIM + kb * 32 + kh * 16 + 8);
      for (int i = 0; i < 8; ++i) { aq[kb][i] = lo[i]; aq[kb][8 + i] = hi[i]; }
    }
  }

  v8f acc[4] = {};
  float mst[8], lst[8];
  for (int v = 0; v < 8; ++v) { mst[v] = -3.0e38f; lst[v] = 0.0f; }

  for (int kc = 0; kc < SEQ / 64; ++kc) {
    const int key0 = kc * 64;
    if (kc + 1 < SEQ / 64) {  // speculative prefetch of next K/V chunk
      __builtin_prefetch(Kp + (size_t)(key0 + 64 + lane) * HDIM, 0, 0);
      __builtin_prefetch(Kp + (size_t)(key0 + 96 + lane) * HDIM, 0, 0);
      __builtin_prefetch(Vp + (size_t)lane * SEQ + key0 + 64, 0, 0);
      __builtin_prefetch(Vp + (size_t)(lane + 32) * SEQ + key0 + 64, 0, 0);
    }
    v8f s[4];
    for (int nt = 0; nt < 4; ++nt) {
      const __bf16* kp = Kp + (size_t)(key0 + nt * 16 + lm) * HDIM;
      v16bf b0 = *(const v16bf*)(kp + kh * 16);
      v16bf b1 = *(const v16bf*)(kp + 32 + kh * 16);
      v8f z = {};
      z = wmma_bf16(aq[0], b0, z);
      z = wmma_bf16(aq[1], b1, z);
      s[nt] = z;
    }
    for (int v = 0; v < 8; ++v) {
      const float mx = rmax16(fmaxf(fmaxf(s[0][v], s[1][v]),
                                    fmaxf(s[2][v], s[3][v])));
      const float mnew = fmaxf(mst[v], mx);
      const float al   = __expf(mst[v] - mnew);
      mst[v] = mnew;
      const float p0 = __expf(s[0][v] - mnew);
      const float p1 = __expf(s[1][v] - mnew);
      const float p2 = __expf(s[2][v] - mnew);
      const float p3 = __expf(s[3][v] - mnew);
      lst[v] = lst[v] * al + rsum16((p0 + p1) + (p2 + p3));
      s[0][v] = p0; s[1][v] = p1; s[2][v] = p2; s[3][v] = p3;
      acc[0][v] *= al; acc[1][v] *= al; acc[2][v] *= al; acc[3][v] *= al;
    }
    for (int nt = 0; nt < 4; ++nt)
      for (int v = 0; v < 8; v += 2) {
        v2bf q = pk2(s[nt][v], s[nt][v + 1]);
        Ps[wvb][v + 8 * kh][nt * 16 + lm]     = q[0];
        Ps[wvb][v + 1 + 8 * kh][nt * 16 + lm] = q[1];
      }
    v16bf ap[2];
    for (int kb = 0; kb < 2; ++kb) {
      v8bf lo = *(const v8bf*)&Ps[wvb][lm][kb * 32 + kh * 16];
      v8bf hi = *(const v8bf*)&Ps[wvb][lm][kb * 32 + kh * 16 + 8];
      for (int i = 0; i < 8; ++i) { ap[kb][i] = lo[i]; ap[kb][8 + i] = hi[i]; }
    }
    for (int j = 0; j < 4; ++j) {
      const __bf16* vp = Vp + (size_t)(j * 16 + lm) * SEQ + key0 + kh * 16;
      v16bf bv0 = *(const v16bf*)(vp);
      v16bf bv1 = *(const v16bf*)(vp + 32);
      acc[j] = wmma_bf16(ap[0], bv0, acc[j]);
      acc[j] = wmma_bf16(ap[1], bv1, acc[j]);
    }
  }

  for (int v = 0; v < 8; ++v) {
    const float inv = 1.0f / lst[v];
    const int t = mt * 16 + v + 8 * kh;
    __bf16* yp = Y + ((size_t)(bb * SEQ + t)) * DMODEL + h * HDIM;
    for (int j = 0; j < 4; ++j)
      yp[j * 16 + lm] = f2bf(acc[j][v] * inv);
  }
}

// ---------------------------------------------------------------------------
// Kernel 3: out = Y @ w_o (bf16 x bf16 -> f32), strided f32 epilogue.
// ---------------------------------------------------------------------------
__global__ __launch_bounds__(256)
void out_gemm_kernel(const __bf16* __restrict__ Yb, const __bf16* __restrict__ Wb,
                     float* __restrict__ Out)
{
  GEMM_PROLOG_AND_LOOP(Yb, Wb, DMODEL)

  for (int sm = 0; sm < 2; ++sm)
    for (int sn = 0; sn < 4; ++sn) {
      const int nn = n0 + wn * 64 + sn * 16 + lm;
      const int mb = m0 + wm * 32 + sm * 16 + 8 * kh;
      float* p = Out + (size_t)mb * DMODEL + nn;
      for (int v = 0; v < 8; ++v) p[(size_t)v * DMODEL] = c[sm][sn][v];
    }
}

// ---------------------------------------------------------------------------
extern "C" void kernel_launch(void* const* d_in, const int* in_sizes, int n_in,
                              void* d_out, int out_size, void* d_ws, size_t ws_size,
                              hipStream_t stream) {
  const float* x     = (const float*)d_in[0];   // [4, 2048, 1024]
  const float* w_qkv = (const float*)d_in[1];   // [1024, 3072]
  const float* w_o   = (const float*)d_in[2];   // [1024, 1024]
  float* out = (float*)d_out;                   // [4, 2048, 1024]

  const size_t per = (size_t)BATCH * NHEAD * SEQ * HDIM;  // 8,388,608 elems
  const size_t PB  = per * 2;                             // 16 MB
  char* ws = (char*)d_ws;
  __bf16* Qh  = (__bf16*)(ws);                 // 16 MB (pre-scaled Q)
  __bf16* Kh  = (__bf16*)(ws + 1 * PB);        // 16 MB
  __bf16* Vt  = (__bf16*)(ws + 2 * PB);        // 16 MB (transposed V)
  __bf16* Y   = (__bf16*)(ws + 3 * PB);        // 16 MB attention output
  __bf16* Xb  = (__bf16*)(ws + 4 * PB);        // 16 MB x in bf16
  __bf16* Wqb = (__bf16*)(ws + 5 * PB);        //  6 MB w_qkv in bf16
  __bf16* Wob = (__bf16*)(ws + 5 * PB + (size_t)DMODEL * 3 * DMODEL * 2); // 2 MB

  const int nx = BATCH * SEQ * DMODEL / 8;       // 1,048,576 groups of 8
  const int nq = DMODEL * 3 * DMODEL / 8;        //   393,216
  const int no = DMODEL * DMODEL / 8;            //   131,072
  cvt_bf16_kernel<<<dim3((nx + 255) / 256), 256, 0, stream>>>(x, Xb, nx);
  cvt_bf16_kernel<<<dim3((nq + 255) / 256), 256, 0, stream>>>(w_qkv, Wqb, nq);
  cvt_bf16_kernel<<<dim3((no + 255) / 256), 256, 0, stream>>>(w_o, Wob, no);

  qkv_gemm_kernel<<<dim3(3 * DMODEL / 128, BATCH * SEQ / 128), 256, 0, stream>>>(
      Xb, Wqb, Qh, Kh, Vt);
  attn_kernel<<<dim3(BATCH * NHEAD * (SEQ / 16) / 8), 256, 0, stream>>>(
      Qh, Kh, Vt, Y);
  out_gemm_kernel<<<dim3(DMODEL / 128, BATCH * SEQ / 128), 256, 0, stream>>>(
      Y, Wob, out);
}
